// _SymbolicPredicateGrounding_75058848465014
// MI455X (gfx1250) — compile-verified, matrix operand order checked by reading, not run on hardware
//
#include <hip/hip_runtime.h>
#include <hip/hip_bf16.h>

typedef __attribute__((ext_vector_type(16))) __bf16 v16bf;
typedef __attribute__((ext_vector_type(8)))  __bf16 v8bf;
typedef __attribute__((ext_vector_type(2)))  __bf16 v2bf;
typedef __attribute__((ext_vector_type(8)))  float  v8f;
typedef int v4i __attribute__((vector_size(16)));   // matches async-LDS builtin pointee

#define N_PRED 29
#define F_DIM  8192
#define H1DIM  256
#define H2DIM  128
#define BATCH  4096

#if __has_builtin(__builtin_amdgcn_global_load_async_to_lds_b128)
#define HAVE_ASYNC 1
#else
#define HAVE_ASYNC 0
#endif

// ---- f32 -> bf16 (RNE) helpers -------------------------------------------
__device__ __forceinline__ unsigned short f2bf(float f) {
  unsigned int u = __float_as_uint(f);
  u += 0x7FFFu + ((u >> 16) & 1u);
  return (unsigned short)(u >> 16);
}

// pack two floats into two bf16 in one u32 (low = x, high = y)
__device__ __forceinline__ unsigned int f2bf2(float x, float y) {
#if __has_builtin(__builtin_amdgcn_cvt_pk_bf16_f32)
  v2bf r = __builtin_amdgcn_cvt_pk_bf16_f32(x, y);
  return __builtin_bit_cast(unsigned int, r);
#else
  return (unsigned int)f2bf(x) | ((unsigned int)f2bf(y) << 16);
#endif
}

// ---- 16B global -> LDS copy (async DMA when available) --------------------
__device__ __forceinline__ void lds_cp16(unsigned short* l, const unsigned short* g) {
#if HAVE_ASYNC
  __builtin_amdgcn_global_load_async_to_lds_b128(
      (__attribute__((address_space(1))) v4i*)g,
      (__attribute__((address_space(3))) v4i*)l, 0, 0);
#else
  *reinterpret_cast<uint4*>(l) = *reinterpret_cast<const uint4*>(g);
#endif
}

__device__ __forceinline__ void wait_async() {
#if HAVE_ASYNC
#if __has_builtin(__builtin_amdgcn_s_wait_asynccnt)
  __builtin_amdgcn_s_wait_asynccnt(0);
#else
  asm volatile("s_wait_asynccnt 0" ::: "memory");
#endif
#endif
}

// Build a 16x32 bf16 WMMA fragment from an LDS row (row-major along K).
// ISA layout: lanes 0-15 hold K = {0..7, 16..23}, lanes 16-31 K = {8..15, 24..31}.
__device__ __forceinline__ v16bf load_frag(const unsigned short* rowPtr, int lane) {
  const int off = (lane & 16) ? 8 : 0;
  v8bf lo = *reinterpret_cast<const v8bf*>(rowPtr + off);
  v8bf hi = *reinterpret_cast<const v8bf*>(rowPtr + off + 16);
  return __builtin_shufflevector(lo, hi, 0,1,2,3,4,5,6,7,8,9,10,11,12,13,14,15);
}

// ---------------------------------------------------------------------------
// Kernel 0: convert features f32 -> bf16 once (A operand reused 29x)
// ---------------------------------------------------------------------------
__global__ __launch_bounds__(256)
void _spg_cvt_features(const float* __restrict__ F, unsigned short* __restrict__ Fb) {
  const size_t i = ((size_t)blockIdx.x * 256 + threadIdx.x) * 8;
  const float4 a = *reinterpret_cast<const float4*>(F + i);
  const float4 b = *reinterpret_cast<const float4*>(F + i + 4);
  uint4 o;
  o.x = f2bf2(a.x, a.y); o.y = f2bf2(a.z, a.w);
  o.z = f2bf2(b.x, b.y); o.w = f2bf2(b.z, b.w);
  *reinterpret_cast<uint4*>(Fb + i) = o;
}

// ---------------------------------------------------------------------------
// Kernel 1: h1[p][m][n] = relu(Fb[m][:] @ W1[p][:][n] + b1[p][n])  (bf16 out)
// Block tile 128x64, BK=64, double-buffered LDS: async A-copy + B global loads
// for tile i+1 are issued before the WMMAs of tile i; one barrier per iter.
// ---------------------------------------------------------------------------
#define K1_BM 128
#define K1_BN 64
#define K1_BK 64
#define K1_LS 72   // 64 + 8 pad halves; 144B row stride (16B-aligned)

__global__ __launch_bounds__(256)
void _spg_gemm1_kernel(const unsigned short* __restrict__ Fb, const float* __restrict__ W1,
                       const float* __restrict__ b1, unsigned short* __restrict__ H1out) {
  __shared__ unsigned short tA[2][K1_BM * K1_LS];
  __shared__ unsigned short tB[2][K1_BN * K1_LS];

  const int tid  = threadIdx.x;
  const int lane = tid & 31;
  const int w    = tid >> 5;
  const int wm   = w >> 1;          // 0..3
  const int wn   = w & 1;           // 0..1
  const int p     = blockIdx.z;
  const int mBase = blockIdx.x * K1_BM;
  const int nBase = blockIdx.y * K1_BN;

  const float* W1p = W1 + (size_t)p * F_DIM * H1DIM;

  // staging thread roles
  const int arow  = tid >> 1;            // 0..127
  const int abase = (tid & 1) * 32;      // halves
  const int kk    = (tid >> 4) * 2;      // 0,2,..,30
  const int n4    = (tid & 15) * 4;      // 0..60

  float4 bx[2][2];                       // B-tile regs: [pass][k-pair]

  auto stageA = [&](unsigned short* dst, int k0) {
    const unsigned short* g = &Fb[(size_t)(mBase + arow) * F_DIM + k0 + abase];
    unsigned short* l = &dst[arow * K1_LS + abase];
    #pragma unroll
    for (int j = 0; j < 4; ++j) lds_cp16(l + j * 8, g + j * 8);
  };
  auto loadB = [&](int k0) {
    #pragma unroll
    for (int pass = 0; pass < 2; ++pass) {
      const int k = kk + pass * 32;
      bx[pass][0] = *reinterpret_cast<const float4*>(&W1p[(size_t)(k0 + k)     * H1DIM + nBase + n4]);
      bx[pass][1] = *reinterpret_cast<const float4*>(&W1p[(size_t)(k0 + k + 1) * H1DIM + nBase + n4]);
    }
  };
  auto storeB = [&](unsigned short* dst) {
    #pragma unroll
    for (int pass = 0; pass < 2; ++pass) {
      const int k = kk + pass * 32;
      *reinterpret_cast<unsigned int*>(&dst[(n4 + 0) * K1_LS + k]) = f2bf2(bx[pass][0].x, bx[pass][1].x);
      *reinterpret_cast<unsigned int*>(&dst[(n4 + 1) * K1_LS + k]) = f2bf2(bx[pass][0].y, bx[pass][1].y);
      *reinterpret_cast<unsigned int*>(&dst[(n4 + 2) * K1_LS + k]) = f2bf2(bx[pass][0].z, bx[pass][1].z);
      *reinterpret_cast<unsigned int*>(&dst[(n4 + 3) * K1_LS + k]) = f2bf2(bx[pass][0].w, bx[pass][1].w);
    }
  };

  v8f acc[2][2] = {};
  const int r16 = lane & 15;

  // prologue: stage tile 0
  stageA(tA[0], 0);
  loadB(0);
  storeB(tB[0]);
  wait_async();
  __syncthreads();

  int cur = 0;
  for (int k0 = 0; k0 < F_DIM; k0 += K1_BK) {
    const int nxt = cur ^ 1;
    const bool hasNext = (k0 + K1_BK) < F_DIM;   // block-uniform

    if (hasNext) { stageA(tA[nxt], k0 + K1_BK); loadB(k0 + K1_BK); }

    const unsigned short* cA = tA[cur];
    const unsigned short* cB = tB[cur];
    #pragma unroll
    for (int ks = 0; ks < K1_BK; ks += 32) {
      v16bf a0  = load_frag(&cA[(wm * 32 +      r16) * K1_LS + ks], lane);
      v16bf a1  = load_frag(&cA[(wm * 32 + 16 + r16) * K1_LS + ks], lane);
      v16bf b0  = load_frag(&cB[(wn * 32 +      r16) * K1_LS + ks], lane);
      v16bf b1v = load_frag(&cB[(wn * 32 + 16 + r16) * K1_LS + ks], lane);
      acc[0][0] = __builtin_amdgcn_wmma_f32_16x16x32_bf16(false, a0, false, b0,  (short)0, acc[0][0], false, false);
      acc[0][1] = __builtin_amdgcn_wmma_f32_16x16x32_bf16(false, a0, false, b1v, (short)0, acc[0][1], false, false);
      acc[1][0] = __builtin_amdgcn_wmma_f32_16x16x32_bf16(false, a1, false, b0,  (short)0, acc[1][0], false, false);
      acc[1][1] = __builtin_amdgcn_wmma_f32_16x16x32_bf16(false, a1, false, b1v, (short)0, acc[1][1], false, false);
    }

    if (hasNext) { storeB(tB[nxt]); }
    wait_async();
    __syncthreads();
    cur = nxt;
  }

  // ---- epilogue: +b1, relu, store bf16 h1[p][m][n] ----
  const int hi8 = (lane >> 4) * 8;   // C layout: lanes 16-31 hold M = 8..15
  #pragma unroll
  for (int mi = 0; mi < 2; ++mi) {
    #pragma unroll
    for (int ni = 0; ni < 2; ++ni) {
      const int n_g = nBase + wn * 32 + ni * 16 + r16;
      const float bias = b1[p * H1DIM + n_g];
      #pragma unroll
      for (int r = 0; r < 8; ++r) {
        const int m_g = mBase + wm * 32 + mi * 16 + r + hi8;
        float v = acc[mi][ni][r] + bias;
        v = v > 0.f ? v : 0.f;
        H1out[((size_t)p * BATCH + m_g) * H1DIM + n_g] = f2bf(v);
      }
    }
  }
}

// ---------------------------------------------------------------------------
// Kernel 2 (fused GEMM2 + head): h2 = relu(h1 @ W2 + b2);
//   out[b][p] = sigmoid(h2 . W3[p] + b3[p])
// Block tile 128x128 (BN == H2), BK=64, double-buffered like kernel 1.
// ---------------------------------------------------------------------------
#define K2_BM 128
#define K2_BK 64
#define K2_LS 72

__global__ __launch_bounds__(256)
void _spg_gemm2_fused_kernel(const unsigned short* __restrict__ H1in,
                             const float* __restrict__ W2, const float* __restrict__ b2,
                             const float* __restrict__ W3, const float* __restrict__ b3,
                             float* __restrict__ out) {
  __shared__ unsigned short tA[2][K2_BM * K2_LS];
  __shared__ unsigned short tB[2][H2DIM * K2_LS];
  __shared__ float rowacc[K2_BM];

  const int tid  = threadIdx.x;
  const int lane = tid & 31;
  const int w    = tid >> 5;
  const int wm   = w >> 1;          // 0..3
  const int wn   = w & 1;           // 0..1 (64 N each)
  const int p     = blockIdx.y;
  const int mBase = blockIdx.x * K2_BM;

  const unsigned short* A = H1in + (size_t)p * BATCH * H1DIM;
  const float* W2p = W2 + (size_t)p * H1DIM * H2DIM;

  const int arow  = tid >> 1;
  const int abase = (tid & 1) * 32;
  const int kk    = (tid >> 5) * 2;      // 0,2,..,14
  const int n4    = (tid & 31) * 4;      // 0..124

  float4 bx[4][2];

  auto stageA = [&](unsigned short* dst, int k0) {
    const unsigned short* g = &A[(size_t)(mBase + arow) * H1DIM + k0 + abase];
    unsigned short* l = &dst[arow * K2_LS + abase];
    #pragma unroll
    for (int j = 0; j < 4; ++j) lds_cp16(l + j * 8, g + j * 8);
  };
  auto loadB = [&](int k0) {
    #pragma unroll
    for (int pass = 0; pass < 4; ++pass) {
      const int k = kk + pass * 16;
      bx[pass][0] = *reinterpret_cast<const float4*>(&W2p[(size_t)(k0 + k)     * H2DIM + n4]);
      bx[pass][1] = *reinterpret_cast<const float4*>(&W2p[(size_t)(k0 + k + 1) * H2DIM + n4]);
    }
  };
  auto storeB = [&](unsigned short* dst) {
    #pragma unroll
    for (int pass = 0; pass < 4; ++pass) {
      const int k = kk + pass * 16;
      *reinterpret_cast<unsigned int*>(&dst[(n4 + 0) * K2_LS + k]) = f2bf2(bx[pass][0].x, bx[pass][1].x);
      *reinterpret_cast<unsigned int*>(&dst[(n4 + 1) * K2_LS + k]) = f2bf2(bx[pass][0].y, bx[pass][1].y);
      *reinterpret_cast<unsigned int*>(&dst[(n4 + 2) * K2_LS + k]) = f2bf2(bx[pass][0].z, bx[pass][1].z);
      *reinterpret_cast<unsigned int*>(&dst[(n4 + 3) * K2_LS + k]) = f2bf2(bx[pass][0].w, bx[pass][1].w);
    }
  };

  v8f acc[2][4] = {};
  const int r16 = lane & 15;

  stageA(tA[0], 0);
  loadB(0);
  storeB(tB[0]);
  wait_async();
  __syncthreads();

  int cur = 0;
  for (int k0 = 0; k0 < H1DIM; k0 += K2_BK) {
    const int nxt = cur ^ 1;
    const bool hasNext = (k0 + K2_BK) < H1DIM;   // block-uniform

    if (hasNext) { stageA(tA[nxt], k0 + K2_BK); loadB(k0 + K2_BK); }

    const unsigned short* cA = tA[cur];
    const unsigned short* cB = tB[cur];
    #pragma unroll
    for (int ks = 0; ks < K2_BK; ks += 32) {
      v16bf a0 = load_frag(&cA[(wm * 32 +      r16) * K2_LS + ks], lane);
      v16bf a1 = load_frag(&cA[(wm * 32 + 16 + r16) * K2_LS + ks], lane);
      #pragma unroll
      for (int ni = 0; ni < 4; ++ni) {
        v16bf bf = load_frag(&cB[(wn * 64 + ni * 16 + r16) * K2_LS + ks], lane);
        acc[0][ni] = __builtin_amdgcn_wmma_f32_16x16x32_bf16(false, a0, false, bf, (short)0, acc[0][ni], false, false);
        acc[1][ni] = __builtin_amdgcn_wmma_f32_16x16x32_bf16(false, a1, false, bf, (short)0, acc[1][ni], false, false);
      }
    }

    if (hasNext) { storeB(tB[nxt]); }
    wait_async();
    __syncthreads();
    cur = nxt;
  }

  // ---- fused head: relu(+b2) * W3, per-row ds_add_f32 reduction, sigmoid ----
  if (tid < K2_BM) rowacc[tid] = 0.f;
  __syncthreads();

  const int hi8 = (lane >> 4) * 8;
  #pragma unroll
  for (int mi = 0; mi < 2; ++mi) {
    #pragma unroll
    for (int ni = 0; ni < 4; ++ni) {
      const int n_g = wn * 64 + ni * 16 + r16;
      const float bias = b2[p * H2DIM + n_g];
      const float w3v  = W3[p * H2DIM + n_g];
      #pragma unroll
      for (int r = 0; r < 8; ++r) {
        const int m_l = wm * 32 + mi * 16 + r + hi8;
        float v = acc[mi][ni][r] + bias;
        v = v > 0.f ? v : 0.f;
        atomicAdd(&rowacc[m_l], v * w3v);
      }
    }
  }
  __syncthreads();

  if (tid < K2_BM) {
    const float x = rowacc[tid] + b3[p];
    out[(size_t)(mBase + tid) * N_PRED + p] = 1.f / (1.f + __expf(-x));
  }
}

// ---------------------------------------------------------------------------
extern "C" void kernel_launch(void* const* d_in, const int* in_sizes, int n_in,
                              void* d_out, int out_size, void* d_ws, size_t ws_size,
                              hipStream_t stream) {
  (void)in_sizes; (void)n_in; (void)out_size; (void)ws_size;
  const float* F  = (const float*)d_in[0];   // [4096, 8192]
  const float* W1 = (const float*)d_in[1];   // [29, 8192, 256]
  const float* b1 = (const float*)d_in[2];   // [29, 256]
  const float* W2 = (const float*)d_in[3];   // [29, 256, 128]
  const float* b2 = (const float*)d_in[4];   // [29, 128]
  const float* W3 = (const float*)d_in[5];   // [29, 128]
  const float* b3 = (const float*)d_in[6];   // [29]
  float* out = (float*)d_out;                // [4096, 29]

  // ws layout: [0, 64MB)  features bf16   [4096][8192]
  //            [64MB, +58MB) h1 bf16      [29][4096][256]
  unsigned short* Fb   = (unsigned short*)d_ws;
  unsigned short* H1ws = Fb + (size_t)BATCH * F_DIM;

  _spg_cvt_features<<<(BATCH * (size_t)F_DIM) / (256 * 8), 256, 0, stream>>>(F, Fb);

  dim3 g1(BATCH / K1_BM, H1DIM / K1_BN, N_PRED);   // (32, 4, 29)
  _spg_gemm1_kernel<<<g1, 256, 0, stream>>>(Fb, W1, b1, H1ws);

  dim3 g2(BATCH / K2_BM, N_PRED);                  // (32, 29)
  _spg_gemm2_fused_kernel<<<g2, 256, 0, stream>>>(H1ws, W2, b2, W3, b3, out);
}
